// Net_GCN_Eigen_GlobalPooling_V2_56727928046284
// MI455X (gfx1250) — matrix operand order repaired and checked by reading, not compiled, not run on hardware
//
#include <hip/hip_runtime.h>

typedef __attribute__((ext_vector_type(16))) _Float16 v16h;
typedef __attribute__((ext_vector_type(8)))  _Float16 h8;
typedef __attribute__((ext_vector_type(8)))  float    v8f;

#define NNODES 160000
#define NBATCH 8
#define NEDGE  1280000
#define K1CAP  40000   /* 5000*8 */
#define K2CAP  10000   /* 1250*8 */
#define K3CAP  2496    /* 312*8  */
#define HCAP   (1u<<22)

// ---------------------------------------------------------------- utilities
__global__ void fill_f32_kernel(float* p, float v, long long n) {
  long long i = (long long)blockIdx.x * blockDim.x + threadIdx.x;
  if (i < n) p[i] = v;
}
__global__ void fill_i32_kernel(int* p, int v, long long n) {
  long long i = (long long)blockIdx.x * blockDim.x + threadIdx.x;
  if (i < n) p[i] = v;
}
__global__ void set_meta_kernel(int* meta) {
  int t = threadIdx.x;
  if (t < 16) meta[t] = (t == 0) ? NNODES : (t == 4) ? NEDGE : 0;
}

__device__ inline void atomMaxF(float* a, float v) {
  int* ia = (int*)a;
  int cur = __float_as_int(*a);
  while (v > __int_as_float(cur)) {
    int old = atomicCAS(ia, cur, __float_as_int(v));
    if (old == cur) break;
    cur = old;
  }
}

// ---------------------------------------------------------------- pooling index build
__global__ void code1_kernel(const float* __restrict__ x, const int* __restrict__ batch,
                             int* code, int* flags) {
  int i = blockIdx.x * blockDim.x + threadIdx.x;
  if (i >= NNODES) return;
  int c0 = (int)x[(long long)i * 42 + 39];
  int cd = c0 + batch[i] * 5000;
  code[i] = cd;
  flags[cd] = 1;
}

__global__ void code_next_kernel(const int* __restrict__ nc, int stride,
                                 const int* __restrict__ nb,
                                 const int* __restrict__ nPtr, int mcap,
                                 int* code, int* flags) {
  int n = nPtr[0];
  int i = blockIdx.x * blockDim.x + threadIdx.x;
  if (i >= n) return;
  int cd = nc[i * stride] + nb[i] * mcap;
  code[i] = cd;
  flags[cd] = 1;
}

// single-block exclusive scan over flags (K <= 40000)
__global__ void scan_kernel(const int* __restrict__ flags, int K, int* rank, int* nOut) {
  __shared__ int sdata[1024];
  __shared__ int carry;
  if (threadIdx.x == 0) carry = 0;
  __syncthreads();
  for (int base = 0; base < K; base += 1024) {
    int i = base + (int)threadIdx.x;
    int v = (i < K) ? flags[i] : 0;
    sdata[threadIdx.x] = v;
    __syncthreads();
    for (int off = 1; off < 1024; off <<= 1) {
      int t = (threadIdx.x >= (unsigned)off) ? sdata[threadIdx.x - off] : 0;
      __syncthreads();
      sdata[threadIdx.x] += t;
      __syncthreads();
    }
    if (i < K) rank[i] = carry + sdata[threadIdx.x] - v;
    __syncthreads();
    if (threadIdx.x == 1023) carry += sdata[1023];
    __syncthreads();
  }
  if (threadIdx.x == 0) *nOut = carry;
}

__global__ void build_inv_kernel(const int* __restrict__ code, const int* __restrict__ nPtr,
                                 const int* __restrict__ rank, int* inv) {
  int n = nPtr[0];
  int i = blockIdx.x * blockDim.x + threadIdx.x;
  if (i < n) inv[i] = rank[code[i]];
}

__global__ void attrs1_kernel(const float* __restrict__ x, const int* __restrict__ batch,
                              const int* __restrict__ inv, int* nc1, int* nb1) {
  int i = blockIdx.x * blockDim.x + threadIdx.x;
  if (i >= NNODES) return;
  int j = inv[i];
  atomicMax(&nc1[j * 2 + 0], (int)x[(long long)i * 42 + 40]);
  atomicMax(&nc1[j * 2 + 1], (int)x[(long long)i * 42 + 41]);
  nb1[j] = batch[i];
}

__global__ void pool_attr2_kernel(const int* __restrict__ nc1, const int* __restrict__ nb1,
                                  const int* __restrict__ inv, const int* __restrict__ nPtr,
                                  int* nc2, int* nb2) {
  int n = nPtr[0];
  int i = blockIdx.x * blockDim.x + threadIdx.x;
  if (i >= n) return;
  int j = inv[i];
  atomicMax(&nc2[j], nc1[i * 2 + 1]);
  nb2[j] = nb1[i];
}

__global__ void pool_attr3_kernel(const int* __restrict__ nb2, const int* __restrict__ inv,
                                  const int* __restrict__ nPtr, int* nb3) {
  int n = nPtr[0];
  int i = blockIdx.x * blockDim.x + threadIdx.x;
  if (i >= n) return;
  nb3[inv[i]] = nb2[i];
}

// deduplicate pooled edges via atomicCAS hash set; code = e0*n + e1 < n^2 < 2^31
__global__ void edge_pool_kernel(const int* __restrict__ src, const int* __restrict__ dst,
                                 const int* __restrict__ eCntPtr,
                                 const int* __restrict__ inv, const int* __restrict__ nPtr,
                                 unsigned* hash, int* outS, int* outD, int* outCnt) {
  int e = blockIdx.x * blockDim.x + threadIdx.x;
  if (e >= eCntPtr[0]) return;
  unsigned n = (unsigned)nPtr[0];
  int e0 = inv[src[e]];
  int e1 = inv[dst[e]];
  if (e0 == e1) return;
  unsigned codev = (unsigned)e0 * n + (unsigned)e1;
  unsigned h = (codev * 2654435761u) & (HCAP - 1u);
  for (;;) {
    unsigned prev = atomicCAS(&hash[h], 0xFFFFFFFFu, codev);
    if (prev == 0xFFFFFFFFu) {
      int o = atomicAdd(outCnt, 1);
      outS[o] = e0;
      outD[o] = e1;
      return;
    }
    if (prev == codev) return;
    h = (h + 1u) & (HCAP - 1u);
  }
}

// ---------------------------------------------------------------- GEMM staging
// fp32 -> fp16 activation convert with K padded to multiple of 32 (pad cols = 0)
__global__ void cvt_x_kernel(const float* __restrict__ X, int ldx, int fin, int ldh,
                             const int* __restrict__ nPtr, _Float16* __restrict__ Xh) {
  int n = nPtr[0];
  long long idx = (long long)blockIdx.x * blockDim.x + threadIdx.x;
  if (idx >= (long long)n * ldh) return;
  int i = (int)(idx / ldh);
  int k = (int)(idx % ldh);
  float v = (k < fin) ? X[(long long)i * ldx + k] : 0.0f;
  Xh[idx] = (_Float16)v;
}

// pack W (fin x fout, fp32) into WMMA B-operand tiles:
// Bp[((kt*tilesN + nt)*32 + lane)*16 + h], lane = (kin>=16)*16 + (col&15), h = kin&15
__global__ void pack_w_kernel(const float* __restrict__ W, int fin, int finPad, int fout,
                              _Float16* __restrict__ Bp) {
  int idx = blockIdx.x * blockDim.x + threadIdx.x;
  if (idx >= finPad * fout) return;
  int k = idx / fout;
  int col = idx % fout;
  float v = (k < fin) ? W[(long long)k * fout + col] : 0.0f;
  int kt = k >> 5, kin = k & 31;
  int nt = col >> 4, nc = col & 15;
  int lane = ((kin >= 16) ? 16 : 0) + nc;
  int h = kin & 15;
  int tilesN = fout >> 4;
  Bp[(((long long)kt * tilesN + nt) * 32 + lane) * 16 + h] = (_Float16)v;
}

// ---------------------------------------------------------------- WMMA GEMM: Y = Xh @ Bp
// one 16x32 output slab per wave32; branch-free b128-fed inner loop, 2 wmma/iter
__global__ void gemm_wmma_kernel(const _Float16* __restrict__ Xh, int ldh,
                                 const _Float16* __restrict__ Bp,
                                 const int* __restrict__ nPtr,
                                 int finPad, int fout,
                                 float* __restrict__ Y) {
  int n = nPtr[0];
  int lane = threadIdx.x & 31;
  int wave = threadIdx.x >> 5;
  int tilesN = fout >> 4;
  int tilesN2 = fout >> 5;
  int tile = blockIdx.x * (blockDim.x >> 5) + wave;
  int tY = tile / tilesN2;
  int tN2 = tile - tY * tilesN2;
  int mBase = tY << 4;
  if (mBase >= n) return;                      // wave-uniform, EXEC stays full

  int r = lane & 15;
  int hs = lane >> 4;
  const _Float16* aPtr = Xh + (long long)(mBase + r) * ldh + hs * 16;
  const _Float16* bPtr = Bp + ((long long)(tN2 * 2) * 32 + lane) * 16;
  long long bStep = (long long)tilesN * 32 * 16;  // halves per k-tile

  v8f acc0 = {}, acc1 = {};
  int nkt = finPad >> 5;
  for (int kt = 0; kt < nkt; ++kt) {
    h8 a0 = *(const h8*)(aPtr);
    h8 a1 = *(const h8*)(aPtr + 8);
    v16h a = __builtin_shufflevector(a0, a1, 0, 1, 2, 3, 4, 5, 6, 7,
                                     8, 9, 10, 11, 12, 13, 14, 15);
    v16h b0 = *(const v16h*)(bPtr);
    v16h b1 = *(const v16h*)(bPtr + 32 * 16);
    acc0 = __builtin_amdgcn_wmma_f32_16x16x32_f16(false, a, false, b0, (short)0, acc0,
                                                  false, false);
    acc1 = __builtin_amdgcn_wmma_f32_16x16x32_f16(false, a, false, b1, (short)0, acc1,
                                                  false, false);
    aPtr += 32;
    bPtr += bStep;
  }
  int col0 = (tN2 << 5) + r;
  int mOff = hs << 3;
#pragma unroll
  for (int rr = 0; rr < 8; ++rr) {
    int m = mBase + mOff + rr;
    if (m < n) {
      Y[(long long)m * fout + col0] = acc0[rr];
      Y[(long long)m * fout + col0 + 16] = acc1[rr];
    }
  }
}

// ---------------------------------------------------------------- GCN message passing
__global__ void scatter_kernel(const float* __restrict__ y, const int* __restrict__ es,
                               const int* __restrict__ ed, const int* __restrict__ eCntPtr,
                               int f, float* agg) {
  long long idx = (long long)blockIdx.x * blockDim.x + threadIdx.x;
  long long tot = (long long)eCntPtr[0] * f;
  if (idx >= tot) return;
  int e = (int)(idx / f);
  int c = (int)(idx % f);
  atomicAdd(&agg[(long long)ed[e] * f + c], y[(long long)es[e] * f + c]);
}

__global__ void bn_stats_kernel(const float* __restrict__ agg, const float* __restrict__ bias,
                                const int* __restrict__ nPtr, int f, float* sums) {
  int c = threadIdx.x;                          // blockDim.x == f
  int n = nPtr[0];
  float bb = bias[c];
  float s = 0.f, s2 = 0.f;
  for (int i = blockIdx.x; i < n; i += gridDim.x) {
    float v = fmaxf(agg[(long long)i * f + c] + bb, 0.f);
    s += v;
    s2 += v * v;
  }
  atomicAdd(&sums[c], s);
  atomicAdd(&sums[f + c], s2);
}

__global__ void bn_norm_kernel(const float* __restrict__ agg, const float* __restrict__ bias,
                               const float* __restrict__ g, const float* __restrict__ bt,
                               const float* __restrict__ sums, const int* __restrict__ nPtr,
                               int f, float* xout) {
  int n = nPtr[0];
  long long idx = (long long)blockIdx.x * blockDim.x + threadIdx.x;
  if (idx >= (long long)n * f) return;
  int c = (int)(idx % f);
  float v = fmaxf(agg[idx] + bias[c], 0.f);
  float inv_n = 1.f / (float)n;
  float mu = sums[c] * inv_n;
  float var = sums[f + c] * inv_n - mu * mu;
  xout[idx] = g[c] * (v - mu) * rsqrtf(var + 1e-5f) + bt[c];
}

__global__ void segmax_kernel(const float* __restrict__ xin, const int* __restrict__ inv,
                              const int* __restrict__ nPtr, int f, float* xout) {
  int n = nPtr[0];
  long long idx = (long long)blockIdx.x * blockDim.x + threadIdx.x;
  if (idx >= (long long)n * f) return;
  int i = (int)(idx / f);
  int c = (int)(idx % f);
  atomMaxF(&xout[(long long)inv[i] * f + c], xin[idx]);
}

__global__ void segmean_kernel(const float* __restrict__ x, const int* __restrict__ bvec,
                               const int* __restrict__ nPtr, int f, int colOff,
                               float* hsum, float* cnt) {
  int n = nPtr[0];
  long long idx = (long long)blockIdx.x * blockDim.x + threadIdx.x;
  if (idx >= (long long)n * f) return;
  int i = (int)(idx / f);
  int c = (int)(idx % f);
  int b = bvec[i];
  atomicAdd(&hsum[b * 576 + colOff + c], x[idx]);
  if (c == 0) atomicAdd(&cnt[b], 1.0f);
}

// ---------------------------------------------------------------- MLP head (8x576 -> 8x10)
__global__ void fc_kernel(const float* __restrict__ hsum, const float* __restrict__ cnt,
                          const float* __restrict__ W1, const float* __restrict__ b1,
                          const float* __restrict__ W2, const float* __restrict__ b2,
                          float* out) {
  __shared__ float h[8 * 576];
  __shared__ float h1[8 * 256];
  for (int idx = threadIdx.x; idx < 8 * 576; idx += blockDim.x) {
    int b = idx / 576, c = idx % 576;
    int lev = (c < 64) ? 0 : (c < 192) ? 1 : (c < 320) ? 2 : 3;
    h[idx] = hsum[idx] / cnt[lev * 8 + b];
  }
  __syncthreads();
  for (int idx = threadIdx.x; idx < 8 * 256; idx += blockDim.x) {
    int b = idx / 256, o = idx % 256;
    float s = b1[o];
    for (int k = 0; k < 576; ++k) s += h[b * 576 + k] * W1[k * 256 + o];
    h1[idx] = fmaxf(s, 0.f);
  }
  __syncthreads();
  for (int idx = threadIdx.x; idx < 80; idx += blockDim.x) {
    int b = idx / 10, o = idx % 10;
    float s = b2[o];
    for (int k = 0; k < 256; ++k) s += h1[b * 256 + k] * W2[k * 10 + o];
    out[idx] = s;
  }
}

// ---------------------------------------------------------------- host driver
static inline unsigned nblk(long long n, int bs) { return (unsigned)((n + bs - 1) / bs); }

extern "C" void kernel_launch(void* const* d_in, const int* in_sizes, int n_in,
                              void* d_out, int out_size, void* d_ws, size_t ws_size,
                              hipStream_t stream) {
  (void)in_sizes; (void)n_in; (void)out_size; (void)ws_size;

  const float* x     = (const float*)d_in[0];
  const int*   ei    = (const int*)d_in[1];
  const int*   batch = (const int*)d_in[2];
  const float *W[12], *bias[12], *gam[12], *bet[12];
  for (int l = 0; l < 12; ++l) {
    W[l]    = (const float*)d_in[3 + l * 4 + 0];
    bias[l] = (const float*)d_in[3 + l * 4 + 1];
    gam[l]  = (const float*)d_in[3 + l * 4 + 2];
    bet[l]  = (const float*)d_in[3 + l * 4 + 3];
  }
  const float* fcW1 = (const float*)d_in[51];
  const float* fcb1 = (const float*)d_in[52];
  const float* fcW2 = (const float*)d_in[53];
  const float* fcb2 = (const float*)d_in[54];
  float* out = (float*)d_out;

  // ----- workspace carve -----
  size_t off = 0;
  char* base = (char*)d_ws;
  auto carve = [&](size_t bytes) -> void* {
    void* p = base + off;
    off += (bytes + 255) & ~(size_t)255;
    return p;
  };
  float* xb0  = (float*)carve((size_t)NNODES * 64 * 4);
  float* ybuf = (float*)carve((size_t)NNODES * 64 * 4);
  float* agg  = (float*)carve((size_t)NNODES * 64 * 4);
  float* xb1  = (float*)carve((size_t)K1CAP * 128 * 4);
  float* xb2  = (float*)carve((size_t)K2CAP * 128 * 4);
  float* xb3  = (float*)carve((size_t)K3CAP * 256 * 4);
  _Float16* xh = (_Float16*)carve((size_t)NNODES * 64 * 2);  // max of all levels
  _Float16* bp = (_Float16*)carve((size_t)256 * 256 * 2);
  float* sums = (float*)carve(512 * 4);
  float* hsum = (float*)carve(8 * 576 * 4);
  float* cnt  = (float*)carve(32 * 4);
  int* meta  = (int*)carve(16 * 4);
  int* codeb = (int*)carve((size_t)NNODES * 4);
  int* flags = (int*)carve((size_t)K1CAP * 4);
  int* rank  = (int*)carve((size_t)K1CAP * 4);
  int* inv1  = (int*)carve((size_t)NNODES * 4);
  int* inv2  = (int*)carve((size_t)K1CAP * 4);
  int* inv3  = (int*)carve((size_t)K2CAP * 4);
  int* nb1   = (int*)carve((size_t)K1CAP * 4);
  int* nb2   = (int*)carve((size_t)K2CAP * 4);
  int* nb3   = (int*)carve((size_t)K3CAP * 4);
  int* nc1   = (int*)carve((size_t)K1CAP * 2 * 4);
  int* nc2   = (int*)carve((size_t)K2CAP * 4);
  int* e1s = (int*)carve((size_t)NEDGE * 4);
  int* e1d = (int*)carve((size_t)NEDGE * 4);
  int* e2s = (int*)carve((size_t)NEDGE * 4);
  int* e2d = (int*)carve((size_t)NEDGE * 4);
  int* e3s = (int*)carve((size_t)NEDGE * 4);
  int* e3d = (int*)carve((size_t)NEDGE * 4);
  unsigned* hash = (unsigned*)carve((size_t)HCAP * 4);

  set_meta_kernel<<<1, 16, 0, stream>>>(meta);
  fill_f32_kernel<<<nblk(8 * 576, 256), 256, 0, stream>>>(hsum, 0.f, 8 * 576);
  fill_f32_kernel<<<1, 32, 0, stream>>>(cnt, 0.f, 32);

  // ----- pooling level 1 -----
  fill_i32_kernel<<<nblk(K1CAP, 256), 256, 0, stream>>>(flags, 0, K1CAP);
  code1_kernel<<<nblk(NNODES, 256), 256, 0, stream>>>(x, batch, codeb, flags);
  scan_kernel<<<1, 1024, 0, stream>>>(flags, K1CAP, rank, &meta[1]);
  build_inv_kernel<<<nblk(NNODES, 256), 256, 0, stream>>>(codeb, &meta[0], rank, inv1);
  fill_i32_kernel<<<nblk(K1CAP * 2, 256), 256, 0, stream>>>(nc1, 0, K1CAP * 2);
  attrs1_kernel<<<nblk(NNODES, 256), 256, 0, stream>>>(x, batch, inv1, nc1, nb1);
  fill_i32_kernel<<<nblk(HCAP, 256), 256, 0, stream>>>((int*)hash, -1, HCAP);
  edge_pool_kernel<<<nblk(NEDGE, 256), 256, 0, stream>>>(ei, ei + NEDGE, &meta[4], inv1,
                                                         &meta[1], hash, e1s, e1d, &meta[5]);
  // ----- pooling level 2 -----
  fill_i32_kernel<<<nblk(K2CAP, 256), 256, 0, stream>>>(flags, 0, K2CAP);
  code_next_kernel<<<nblk(K1CAP, 256), 256, 0, stream>>>(nc1, 2, nb1, &meta[1], 1250, codeb, flags);
  scan_kernel<<<1, 1024, 0, stream>>>(flags, K2CAP, rank, &meta[2]);
  build_inv_kernel<<<nblk(K1CAP, 256), 256, 0, stream>>>(codeb, &meta[1], rank, inv2);
  fill_i32_kernel<<<nblk(K2CAP, 256), 256, 0, stream>>>(nc2, 0, K2CAP);
  pool_attr2_kernel<<<nblk(K1CAP, 256), 256, 0, stream>>>(nc1, nb1, inv2, &meta[1], nc2, nb2);
  fill_i32_kernel<<<nblk(HCAP, 256), 256, 0, stream>>>((int*)hash, -1, HCAP);
  edge_pool_kernel<<<nblk(NEDGE, 256), 256, 0, stream>>>(e1s, e1d, &meta[5], inv2,
                                                         &meta[2], hash, e2s, e2d, &meta[6]);
  // ----- pooling level 3 -----
  fill_i32_kernel<<<nblk(K3CAP, 256), 256, 0, stream>>>(flags, 0, K3CAP);
  code_next_kernel<<<nblk(K2CAP, 256), 256, 0, stream>>>(nc2, 1, nb2, &meta[2], 312, codeb, flags);
  scan_kernel<<<1, 1024, 0, stream>>>(flags, K3CAP, rank, &meta[3]);
  build_inv_kernel<<<nblk(K2CAP, 256), 256, 0, stream>>>(codeb, &meta[2], rank, inv3);
  pool_attr3_kernel<<<nblk(K2CAP, 256), 256, 0, stream>>>(nb2, inv3, &meta[2], nb3);
  fill_i32_kernel<<<nblk(HCAP, 256), 256, 0, stream>>>((int*)hash, -1, HCAP);
  edge_pool_kernel<<<nblk(NEDGE, 256), 256, 0, stream>>>(e2s, e2d, &meta[6], inv3,
                                                         &meta[3], hash, e3s, e3d, &meta[7]);

  // ----- one GCN layer -----
  auto layer = [&](const float* xin, int ldx, int fin, int fout, int pl,
                   const int* nPtr, int nMax, const int* es, const int* ed,
                   const int* ecnt, float* xout) {
    int finPad = (fin + 31) & ~31;
    cvt_x_kernel<<<nblk((long long)nMax * finPad, 256), 256, 0, stream>>>(xin, ldx, fin,
                                                                          finPad, nPtr, xh);
    pack_w_kernel<<<nblk(finPad * fout, 256), 256, 0, stream>>>(W[pl], fin, finPad, fout, bp);
    int tiles = (fout / 32) * ((nMax + 15) / 16);
    gemm_wmma_kernel<<<nblk(tiles, 4), 128, 0, stream>>>(xh, finPad, bp, nPtr, finPad, fout, ybuf);
    fill_f32_kernel<<<nblk((long long)nMax * fout, 256), 256, 0, stream>>>(agg, 0.f,
                                                                           (long long)nMax * fout);
    fill_f32_kernel<<<2, 256, 0, stream>>>(sums, 0.f, 512);
    scatter_kernel<<<nblk((long long)NEDGE * fout, 256), 256, 0, stream>>>(ybuf, es, ed, ecnt,
                                                                           fout, agg);
    bn_stats_kernel<<<256, fout, 0, stream>>>(agg, bias[pl], nPtr, fout, sums);
    bn_norm_kernel<<<nblk((long long)nMax * fout, 256), 256, 0, stream>>>(
        agg, bias[pl], gam[pl], bet[pl], sums, nPtr, fout, xout);
  };

  // level 0: feats(39) -> 64 -> 64 -> 64
  layer(x,   42, 39, 64, 0, &meta[0], NNODES, ei, ei + NEDGE, &meta[4], xb0);
  layer(xb0, 64, 64, 64, 1, &meta[0], NNODES, ei, ei + NEDGE, &meta[4], xb0);
  layer(xb0, 64, 64, 64, 2, &meta[0], NNODES, ei, ei + NEDGE, &meta[4], xb0);
  segmean_kernel<<<nblk((long long)NNODES * 64, 256), 256, 0, stream>>>(xb0, batch, &meta[0],
                                                                        64, 0, hsum, cnt + 0);
  fill_f32_kernel<<<nblk((long long)K1CAP * 64, 256), 256, 0, stream>>>(xb1, -3.0e38f,
                                                                        (long long)K1CAP * 64);
  segmax_kernel<<<nblk((long long)NNODES * 64, 256), 256, 0, stream>>>(xb0, inv1, &meta[0], 64, xb1);

  // level 1: 64 -> 128 -> 128 -> 128
  layer(xb1, 64,  64,  128, 3, &meta[1], K1CAP, e1s, e1d, &meta[5], xb1);
  layer(xb1, 128, 128, 128, 4, &meta[1], K1CAP, e1s, e1d, &meta[5], xb1);
  layer(xb1, 128, 128, 128, 5, &meta[1], K1CAP, e1s, e1d, &meta[5], xb1);
  segmean_kernel<<<nblk((long long)K1CAP * 128, 256), 256, 0, stream>>>(xb1, nb1, &meta[1],
                                                                        128, 64, hsum, cnt + 8);
  fill_f32_kernel<<<nblk((long long)K2CAP * 128, 256), 256, 0, stream>>>(xb2, -3.0e38f,
                                                                         (long long)K2CAP * 128);
  segmax_kernel<<<nblk((long long)K1CAP * 128, 256), 256, 0, stream>>>(xb1, inv2, &meta[1], 128, xb2);

  // level 2: 128 -> 128 -> 128 -> 128
  layer(xb2, 128, 128, 128, 6, &meta[2], K2CAP, e2s, e2d, &meta[6], xb2);
  layer(xb2, 128, 128, 128, 7, &meta[2], K2CAP, e2s, e2d, &meta[6], xb2);
  layer(xb2, 128, 128, 128, 8, &meta[2], K2CAP, e2s, e2d, &meta[6], xb2);
  segmean_kernel<<<nblk((long long)K2CAP * 128, 256), 256, 0, stream>>>(xb2, nb2, &meta[2],
                                                                        128, 192, hsum, cnt + 16);
  fill_f32_kernel<<<nblk((long long)K3CAP * 128, 256), 256, 0, stream>>>(xb3, -3.0e38f,
                                                                         (long long)K3CAP * 128);
  segmax_kernel<<<nblk((long long)K2CAP * 128, 256), 256, 0, stream>>>(xb2, inv3, &meta[2], 128, xb3);

  // level 3: 128 -> 256 -> 256 -> 256
  layer(xb3, 128, 128, 256, 9,  &meta[3], K3CAP, e3s, e3d, &meta[7], xb3);
  layer(xb3, 256, 256, 256, 10, &meta[3], K3CAP, e3s, e3d, &meta[7], xb3);
  layer(xb3, 256, 256, 256, 11, &meta[3], K3CAP, e3s, e3d, &meta[7], xb3);
  segmean_kernel<<<nblk((long long)K3CAP * 256, 256), 256, 0, stream>>>(xb3, nb3, &meta[3],
                                                                        256, 320, hsum, cnt + 24);

  fc_kernel<<<1, 256, 0, stream>>>(hsum, cnt, fcW1, fcb1, fcW2, fcb2, out);
}
